// NoConvAMPClassifier_4148938408895
// MI455X (gfx1250) — matrix operand order repaired and checked by reading, not compiled, not run on hardware
//
#include <hip/hip_runtime.h>
#include <hip/hip_bf16.h>
#include <math.h>

#define VOCAB 21
#define EMB   128
#define H1    64
#define H2    100
#define H2P   112    // H2 padded to 7*16
#define H2K   128    // K-dim pad for h2 (4 chunks of 32)
#define TT    500
#define G1    256    // 4*H1
#define G2P   448    // 4*H2P

typedef __attribute__((ext_vector_type(16))) __bf16 v16bf;
typedef __attribute__((ext_vector_type(8)))  __bf16 v8bf;
typedef __attribute__((ext_vector_type(8)))  float  v8f;

// ---- fast transcendentals: single-instruction TRANS32 ops (co-execute with WMMA) ----
__device__ __forceinline__ float fast_tanh(float x) {
#if __has_builtin(__builtin_amdgcn_tanhf)
    return __builtin_amdgcn_tanhf(x);            // v_tanh_f32
#else
    float e = __expf(-2.0f * x);                 // v_exp_f32
    return 1.0f - 2.0f * __builtin_amdgcn_rcpf(1.0f + e);
#endif
}
__device__ __forceinline__ float sigm(float x) {
    return __builtin_amdgcn_rcpf(1.0f + __expf(-x));  // v_exp_f32 + v_rcp_f32
}

// A-matrix 16x32 bf16 fragment: lane holds row (lane&15), K = klo..klo+7 and klo+16..klo+23
__device__ __forceinline__ v16bf ld_fragA(const __bf16* p, int klo) {
    v8bf lo = *(const v8bf*)(p + klo);
    v8bf hi = *(const v8bf*)(p + klo + 16);
    v16bf r;
#pragma unroll
    for (int i = 0; i < 8; ++i) { r[i] = lo[i]; r[8 + i] = hi[i]; }
    return r;
}

// B-matrix 32x16 bf16 fragment from row-major W (row = gate col): lane half holds K = 8h..8h+7 and 8h+16..8h+23
__device__ __forceinline__ v16bf ld_fragB(const __bf16* p, int khalf) {
    const __bf16* q = p + (khalf << 3);
    v8bf lo = *(const v8bf*)(q);
    v8bf hi = *(const v8bf*)(q + 16);
    v16bf r;
#pragma unroll
    for (int i = 0; i < 8; ++i) { r[i] = lo[i]; r[8 + i] = hi[i]; }
    return r;
}

__device__ __forceinline__ v8f wmma_bf16(v16bf a, v16bf b, v8f c) {
    return __builtin_amdgcn_wmma_f32_16x16x32_bf16(false, a, false, b, (short)0, c, false, false);
}

__device__ __forceinline__ v8f splat8(float x) {
    v8f r;
#pragma unroll
    for (int i = 0; i < 8; ++i) r[i] = x;
    return r;
}

__global__ __launch_bounds__(128) void lstm_fused_kernel(
    const int* __restrict__ x_idx, const float* __restrict__ emb,
    const float* __restrict__ Wih1, const float* __restrict__ Whh1,
    const float* __restrict__ bih1, const float* __restrict__ bhh1,
    const float* __restrict__ Wih2, const float* __restrict__ Whh2,
    const float* __restrict__ bih2, const float* __restrict__ bhh2,
    const float* __restrict__ fcw, const float* __restrict__ fcb,
    float* __restrict__ out)
{
    __shared__ __align__(16) __bf16 emb_s [VOCAB * EMB];
    __shared__ __align__(16) __bf16 wih1_s[G1 * EMB];
    __shared__ __align__(16) __bf16 whh1_s[G1 * H1];
    __shared__ __align__(16) __bf16 wih2_s[G2P * H1];
    __shared__ __align__(16) __bf16 whh2_s[G2P * H2K];
    __shared__ float bias1_s[G1];
    __shared__ float bias2_s[G2P];
    __shared__ __align__(16) __bf16 h1_s[2][16 * H1];
    __shared__ __align__(16) __bf16 x2_s[16 * H1];
    __shared__ __align__(16) __bf16 h2_s[2][16 * H2K];

    const int tid = threadIdx.x;

    // ---------------- one-time preload: global f32 -> LDS bf16 (padded) -----------
    for (int i = tid; i < VOCAB * EMB; i += 128) emb_s[i]  = (__bf16)emb[i];
    for (int i = tid; i < G1 * EMB;   i += 128) wih1_s[i] = (__bf16)Wih1[i];
    for (int i = tid; i < G1 * H1;    i += 128) whh1_s[i] = (__bf16)Whh1[i];
    for (int i = tid; i < G2P * H1; i += 128) {
        int pr = i >> 6, c = i & 63;
        int g = pr / H2P, lr = pr % H2P;
        wih2_s[i] = (lr < H2) ? (__bf16)Wih2[(g * H2 + lr) * H1 + c] : (__bf16)0.0f;
    }
    for (int i = tid; i < G2P * H2K; i += 128) {
        int pr = i >> 7, c = i & 127;
        int g = pr / H2P, lr = pr % H2P;
        whh2_s[i] = (lr < H2 && c < H2) ? (__bf16)Whh2[(g * H2 + lr) * H2 + c] : (__bf16)0.0f;
    }
    for (int i = tid; i < G1; i += 128) bias1_s[i] = bih1[i] + bhh1[i];
    for (int i = tid; i < G2P; i += 128) {
        int g = i / H2P, lr = i % H2P;
        bias2_s[i] = (lr < H2) ? (bih2[g * H2 + lr] + bhh2[g * H2 + lr]) : 0.0f;
    }
    for (int i = tid; i < 16 * H1; i += 128) { h1_s[0][i] = (__bf16)0.0f; h1_s[1][i] = (__bf16)0.0f; }
    for (int i = tid; i < 16 * H2K; i += 128) { h2_s[0][i] = (__bf16)0.0f; h2_s[1][i] = (__bf16)0.0f; }
    __syncthreads();

    const int wv    = tid >> 5;        // wave id 0..3 -> owns hidden-unit tile column
    const int lane  = tid & 31;
    const int col   = lane & 15;       // matrix row (A) / gate column (B) / N (C)
    const int khalf = lane >> 4;
    const int kloA  = khalf << 3;
    const int b0    = blockIdx.x << 4; // 16 batch rows per block

    // ---- register-resident invariants: W_hh1 B-fragments (reused 500x) + biases ----
    v16bf whh1_f[4][2];
    float b1r[4];
#pragma unroll
    for (int g = 0; g < 4; ++g) {
        int n0 = g * H1 + (wv << 4);
        b1r[g] = bias1_s[n0 + col];
        const __bf16* hr = whh1_s + (n0 + col) * H1;
#pragma unroll
        for (int kc = 0; kc < 2; ++kc) whh1_f[g][kc] = ld_fragB(hr + kc * 32, khalf);
    }
    float b2r[2][4];
#pragma unroll
    for (int ui = 0; ui < 2; ++ui) {
        int uc = wv + (ui << 2);
        if (uc < 7) {
#pragma unroll
            for (int g = 0; g < 4; ++g) b2r[ui][g] = bias2_s[g * H2P + (uc << 4) + col];
        } else {
#pragma unroll
            for (int g = 0; g < 4; ++g) b2r[ui][g] = 0.0f;
        }
    }

    float c1[8], pool[8], c2[2][8];
#pragma unroll
    for (int r = 0; r < 8; ++r) { c1[r] = 0.f; pool[r] = -1e30f; c2[0][r] = 0.f; c2[1][r] = 0.f; }
    int pb1 = 0, pb2 = 0;

    const int* idxrow = x_idx + (b0 + col) * TT;
    int tok = idxrow[0];

    for (int t = 0; t < TT; ++t) {
        int tok_next = (t + 1 < TT) ? idxrow[t + 1] : 0;  // pipeline token fetch

        // ---------------- layer 1 step: gates(16x256) = x*Wih1^T + h*Whh1^T + b ----
        const __bf16* erow = emb_s + tok * EMB;           // emb row 0 is zero => mask
        v16bf ax[4], ah[2];
#pragma unroll
        for (int kc = 0; kc < 4; ++kc) ax[kc] = ld_fragA(erow + kc * 32, kloA);
#pragma unroll
        for (int kc = 0; kc < 2; ++kc) ah[kc] = ld_fragA(h1_s[pb1] + col * H1 + kc * 32, kloA);

        v8f acc[4];
#pragma unroll
        for (int g = 0; g < 4; ++g) {
            int n0 = g * H1 + (wv << 4);
            v8f a = splat8(b1r[g]);
            const __bf16* wr = wih1_s + (n0 + col) * EMB;
#pragma unroll
            for (int kc = 0; kc < 4; ++kc) a = wmma_bf16(ax[kc], ld_fragB(wr + kc * 32, khalf), a);
#pragma unroll
            for (int kc = 0; kc < 2; ++kc) a = wmma_bf16(ah[kc], whh1_f[g][kc], a);
            acc[g] = a;
        }

        const bool doL2 = ((t % 5) == 4);
#pragma unroll
        for (int r = 0; r < 8; ++r) {
            float iv = sigm(acc[0][r]);
            float fv = sigm(acc[1][r]);
            float gv = fast_tanh(acc[2][r]);
            float ov = sigm(acc[3][r]);
            c1[r] = fv * c1[r] + iv * gv;
            float hv = ov * fast_tanh(c1[r]);
            pool[r] = fmaxf(pool[r], hv);
            int mm = r + (khalf << 3);
            h1_s[pb1 ^ 1][mm * H1 + (wv << 4) + col] = (__bf16)hv;
            if (doL2) { x2_s[mm * H1 + (wv << 4) + col] = (__bf16)pool[r]; pool[r] = -1e30f; }
        }
        __syncthreads();
        pb1 ^= 1;
        tok = tok_next;

        // ---------------- layer 2 step (every 5th t): gates(16x448 padded) --------
        if (doL2) {
            v16bf bx[2], bh[4];
#pragma unroll
            for (int kc = 0; kc < 2; ++kc) bx[kc] = ld_fragA(x2_s + col * H1 + kc * 32, kloA);
#pragma unroll
            for (int kc = 0; kc < 4; ++kc) bh[kc] = ld_fragA(h2_s[pb2] + col * H2K + kc * 32, kloA);

            const int nu = (wv < 3) ? 2 : 1;               // 7 unit-tiles over 4 waves
            for (int ui = 0; ui < nu; ++ui) {
                const int uc = wv + (ui << 2);
                v8f a2[4];
#pragma unroll
                for (int g = 0; g < 4; ++g) {
                    int n0 = g * H2P + (uc << 4);
                    v8f a = splat8(b2r[ui][g]);
                    const __bf16* wr = wih2_s + (n0 + col) * H1;
#pragma unroll
                    for (int kc = 0; kc < 2; ++kc) a = wmma_bf16(bx[kc], ld_fragB(wr + kc * 32, khalf), a);
                    const __bf16* hr = whh2_s + (n0 + col) * H2K;
#pragma unroll
                    for (int kc = 0; kc < 4; ++kc) a = wmma_bf16(bh[kc], ld_fragB(hr + kc * 32, khalf), a);
                    a2[g] = a;
                }
#pragma unroll
                for (int r = 0; r < 8; ++r) {
                    float iv = sigm(a2[0][r]);
                    float fv = sigm(a2[1][r]);
                    float gv = fast_tanh(a2[2][r]);
                    float ov = sigm(a2[3][r]);
                    c2[ui][r] = fv * c2[ui][r] + iv * gv;
                    float hv = ov * fast_tanh(c2[ui][r]);
                    int mm = r + (khalf << 3);
                    h2_s[pb2 ^ 1][mm * H2K + (uc << 4) + col] = (__bf16)hv;
                }
            }
            __syncthreads();
            pb2 ^= 1;
        }
    }

    // ---------------- final FC + sigmoid: out[b] = sig(h2 . fc_w + fc_b) ----------
    if (tid < 16) {
        float s = fcb[0];
        const __bf16* hrow = h2_s[pb2] + tid * H2K;
        for (int u = 0; u < H2; ++u) s += (float)hrow[u] * fcw[u];
        out[b0 + tid] = __builtin_amdgcn_rcpf(1.0f + __expf(-s));
    }
}

extern "C" void kernel_launch(void* const* d_in, const int* in_sizes, int n_in,
                              void* d_out, int out_size, void* d_ws, size_t ws_size,
                              hipStream_t stream) {
    const int*   x_idx = (const int*)  d_in[0];
    const float* emb   = (const float*)d_in[1];
    const float* Wih1  = (const float*)d_in[2];
    const float* Whh1  = (const float*)d_in[3];
    const float* bih1  = (const float*)d_in[4];
    const float* bhh1  = (const float*)d_in[5];
    const float* Wih2  = (const float*)d_in[6];
    const float* Whh2  = (const float*)d_in[7];
    const float* bih2  = (const float*)d_in[8];
    const float* bhh2  = (const float*)d_in[9];
    const float* fcw   = (const float*)d_in[10];
    const float* fcb   = (const float*)d_in[11];
    float* out = (float*)d_out;

    hipLaunchKernelGGL(lstm_fused_kernel, dim3(1024 / 16), dim3(128), 0, stream,
                       x_idx, emb, Wih1, Whh1, bih1, bhh1,
                       Wih2, Whh2, bih2, bhh2, fcw, fcb, out);
}
